// GlobalFilterLayer_8589935112
// MI455X (gfx1250) — matrix operand order repaired and checked by reading, not compiled, not run on hardware
//
#include <hip/hip_runtime.h>
#include <math.h>

#define DD 768                 // embed dim = FFT length along D
#define MROWS 12608            // B*N = 64*197, exactly 197*64

typedef __attribute__((ext_vector_type(16))) _Float16 v16h;
typedef __attribute__((ext_vector_type(8)))  float    v8f;
typedef __attribute__((ext_vector_type(8)))  float    f32x8;

// ---------------------------------------------------------------------------
// Kernel 1: hr[m] = Re( IFFT(filter)[m] )
//         = (1/D) * sum_k ( fr[k]*cos(2*pi*k*m/D) - fi[k]*sin(2*pi*k*m/D) )
// 768 threads total; each does a 768-term sum. Phase reduced mod D for accuracy.
// ---------------------------------------------------------------------------
__global__ void hr_kernel(const float* __restrict__ fr,
                          const float* __restrict__ fi,
                          float* __restrict__ hr) {
    int m = blockIdx.x * blockDim.x + threadIdx.x;
    if (m >= DD) return;
    const float w = 6.28318530717958647692f / (float)DD;
    float acc = 0.0f;
    for (int k = 0; k < DD; ++k) {
        int ph = (k * m) % DD;              // keep sincos argument in [0, 2*pi)
        float s, c;
        __sincosf(w * (float)ph, &s, &c);
        acc = fmaf(fr[k],  c, acc);
        acc = fmaf(-fi[k], s, acc);
    }
    hr[m] = acc * (1.0f / (float)DD);
}

// ---------------------------------------------------------------------------
// Kernel 2: build transposed circulant in f16, fragment-friendly layout:
//   Ct[n][k] = hr[(n - k) mod D]   (row-major N x K, K contiguous)
// so a WMMA B-fragment lane (column n, 16 consecutive K) is one 32B load.
// ---------------------------------------------------------------------------
__global__ void ct_kernel(const float* __restrict__ hr,
                          _Float16* __restrict__ ct) {
    int idx = blockIdx.x * blockDim.x + threadIdx.x;   // 768*768 threads
    int n = idx / DD;
    int k = idx - n * DD;
    int m = n - k; if (m < 0) m += DD;
    ct[idx] = (_Float16)hr[m];
}

// ---------------------------------------------------------------------------
// Kernel 3: out = x (12608 x 768, f32) @ Ct^T  via v_wmma_f32_16x16x32_f16.
// Register-blocked 64(M) x 64(N) per wave: 4 M-tiles x 4 N-tiles = 16
// accumulators. B fragments are reused across 4 M-tiles (4x less L2 traffic
// for the circulant), A fragments across 4 N-tiles; 16 WMMAs per K-step give
// the scheduler plenty of compute to overlap with the 24 b128 loads.
// Block = 4 waves spanning 256 N columns over the same 64 M rows (A reads
// shared via WGP$). Grid (197, 3) tiles M and N exactly: no divergence,
// EXEC all-1s as WMMA requires.
// ---------------------------------------------------------------------------
__global__ __launch_bounds__(128)
void gemm_kernel(const float* __restrict__ x,
                 const _Float16* __restrict__ ct,
                 float* __restrict__ out) {
    const int lane = threadIdx.x & 31;
    const int wave = threadIdx.x >> 5;        // 0..3
    const int l15  = lane & 15;               // A row / B col / D col within tile
    const int hsel = lane >> 4;               // half-wave select

    const int m0 = blockIdx.x * 64;                    // block M base (4 tiles of 16)
    const int n0 = blockIdx.y * 256 + wave * 64;       // wave N base  (4 tiles of 16)

    // A: row (m0 + mi*16 + l15); lane covers K = k0 + hsel*16 + [0,16) contiguous
    const float* arow = x + (size_t)(m0 + l15) * DD + hsel * 16;

    // B: column (n0 + t*16 + l15); lane covers K = k0 + hsel*16 + [0,16) contiguous
    const _Float16* bbase = ct + (size_t)(n0 + l15) * DD + hsel * 16;

    v8f acc[4][4] = {};   // [mi][t]

    for (int k0 = 0; k0 < DD; k0 += 32) {
        // ---- 4 B fragments, shared by all 4 M-tiles this K-step ----
        v16h b[4];
#pragma unroll
        for (int t = 0; t < 4; ++t)
            b[t] = *(const v16h*)(bbase + (size_t)t * 16 * DD + k0);

#pragma unroll
        for (int mi = 0; mi < 4; ++mi) {
            // ---- A fragment for M-tile mi: f32 loads + pack to f16 ----
            f32x8 r0 = *(const f32x8*)(arow + (size_t)mi * 16 * DD + k0);
            f32x8 r1 = *(const f32x8*)(arow + (size_t)mi * 16 * DD + k0 + 8);
            v16h a;
#pragma unroll
            for (int j = 0; j < 8; ++j) {
                a[j]     = (_Float16)r0[j];
                a[8 + j] = (_Float16)r1[j];
            }
#pragma unroll
            for (int t = 0; t < 4; ++t)
                acc[mi][t] = __builtin_amdgcn_wmma_f32_16x16x32_f16(
                    false, a, false, b[t], (short)0, acc[mi][t], false, false);
        }
    }

    // ---- Store: D layout is VGPR v -> M = v + 8*hsel, lane&15 -> N ----
#pragma unroll
    for (int mi = 0; mi < 4; ++mi) {
        float* orow = out + (size_t)(m0 + mi * 16 + 8 * hsel) * DD + n0 + l15;
#pragma unroll
        for (int v = 0; v < 8; ++v) {
#pragma unroll
            for (int t = 0; t < 4; ++t)
                orow[(size_t)v * DD + t * 16] = acc[mi][t][v];
        }
    }
}

// ---------------------------------------------------------------------------
// Launch: hr (3 KB ws) -> Ct (1.18 MB ws, offset 4096) -> WMMA GEMM.
// ---------------------------------------------------------------------------
extern "C" void kernel_launch(void* const* d_in, const int* in_sizes, int n_in,
                              void* d_out, int out_size, void* d_ws, size_t ws_size,
                              hipStream_t stream) {
    const float* x  = (const float*)d_in[0];   // (64, 197, 768) f32
    const float* fr = (const float*)d_in[1];   // (768,) f32
    const float* fi = (const float*)d_in[2];   // (768,) f32
    float* out = (float*)d_out;                // (64, 197, 768) f32

    float*    hr = (float*)d_ws;                         // 768 floats
    _Float16* ct = (_Float16*)((char*)d_ws + 4096);      // 768*768 f16

    hr_kernel<<<dim3(3), dim3(256), 0, stream>>>(fr, fi, hr);
    ct_kernel<<<dim3((DD * DD) / 256), dim3(256), 0, stream>>>(hr, ct);
    gemm_kernel<<<dim3(MROWS / 64, DD / 256), dim3(128), 0, stream>>>(x, ct, out);
}
